// RelativeGlobalAttention_31817117729139
// MI455X (gfx1250) — matrix operand order repaired and checked by reading, not compiled, not run on hardware
//
#include <hip/hip_runtime.h>

#if __has_builtin(__builtin_amdgcn_exp2f)
#define EXP2F(x) __builtin_amdgcn_exp2f(x)
#else
#define EXP2F(x) exp2f(x)
#endif

#define S_LEN 2048
#define D_ATT 64
#define BATCH 8
#define QROWS_PER_BLOCK 64
#define WAVES 4
#define NIT (S_LEN / 32)

typedef __attribute__((ext_vector_type(16))) _Float16 v16h;
typedef __attribute__((ext_vector_type(8)))  _Float16 v8h;
typedef __attribute__((ext_vector_type(4)))  _Float16 v4h;
typedef __attribute__((ext_vector_type(8)))  float    v8f;

// 16 consecutive f16 (32B, 16B-aligned) -> WMMA operand registers.
__device__ __forceinline__ v16h ld16(const _Float16* p) {
  const v8h* q = (const v8h*)p;
  v8h lo = q[0], hh = q[1];
  v16h r;
#pragma unroll
  for (int e = 0; e < 8; ++e) { r[e] = lo[e]; r[e + 8] = hh[e]; }
  return r;
}

__device__ __forceinline__ v8f wmma_f16(v16h a, v16h b, v8f c) {
  return __builtin_amdgcn_wmma_f32_16x16x32_f16(false, a, false, b, (short)0, c,
                                                false, false);
}

__device__ __forceinline__ v4h cvt4(float4 f) {
  v4h h = {(_Float16)f.x, (_Float16)f.y, (_Float16)f.z, (_Float16)f.w};
  return h;
}

// Async DMA: copy 16B global -> LDS at (base + OFF) on both sides; tracked by
// ASYNCcnt (CDNA5 GLOBAL_LOAD_ASYNC_TO_LDS_B128: INST_OFFSET is added to both
// the LDS address (VDST) and the global address (VADDR)).
template <int OFF>
__device__ __forceinline__ void async_cp16_off(unsigned lds_byte,
                                               unsigned long long gaddr) {
  asm volatile("global_load_async_to_lds_b128 %0, %1, off offset:%c2"
               :: "v"(lds_byte), "v"(gaddr), "i"(OFF)
               : "memory");
}
__device__ __forceinline__ void wait_async0() {
  asm volatile("s_wait_asynccnt 0x0" ::: "memory");
}

// XOR-mask lane reduction step via DS_SWIZZLE_B32 (immediate pattern:
// {xor[14:10], or[9:5]=0, and[4:0]=0x1F} -> no lane-address VALU).
#if __has_builtin(__builtin_amdgcn_ds_swizzle)
template <int MASK>
__device__ __forceinline__ float swz_xor(float x) {
  return __int_as_float(
      __builtin_amdgcn_ds_swizzle(__float_as_int(x), 0x1F | (MASK << 10)));
}
#else
template <int MASK>
__device__ __forceinline__ float swz_xor(float x) {
  return __shfl_xor(x, MASK, 32);
}
#endif

__device__ __forceinline__ void rowmax16(float (&v)[8]) {
#pragma unroll
  for (int r = 0; r < 8; ++r) v[r] = fmaxf(v[r], swz_xor<1>(v[r]));
#pragma unroll
  for (int r = 0; r < 8; ++r) v[r] = fmaxf(v[r], swz_xor<2>(v[r]));
#pragma unroll
  for (int r = 0; r < 8; ++r) v[r] = fmaxf(v[r], swz_xor<4>(v[r]));
#pragma unroll
  for (int r = 0; r < 8; ++r) v[r] = fmaxf(v[r], swz_xor<8>(v[r]));
}
__device__ __forceinline__ void rowsum16(float (&v)[8]) {
#pragma unroll
  for (int r = 0; r < 8; ++r) v[r] += swz_xor<1>(v[r]);
#pragma unroll
  for (int r = 0; r < 8; ++r) v[r] += swz_xor<2>(v[r]);
#pragma unroll
  for (int r = 0; r < 8; ++r) v[r] += swz_xor<4>(v[r]);
#pragma unroll
  for (int r = 0; r < 8; ++r) v[r] += swz_xor<8>(v[r]);
}

__global__ __launch_bounds__(128) void rga_flash_kernel(
    const float* __restrict__ q, const float* __restrict__ k,
    const float* __restrict__ v, const float* __restrict__ rel,
    float* __restrict__ out) {
  // K DMA landing buffer is UNPADDED so its row stride (256B) matches global;
  // the 4 per-thread chunks then share one base and offset:0/2048/4096/6144.
  __shared__ __align__(16) float    k32_lds[32][64];       // K DMA landing (f32)
  __shared__ __align__(16) _Float16 k16_lds[2][32][72];    // K tile f16 (dbl buf)
  __shared__ __align__(16) _Float16 vT_lds[2][64][40];     // V^T tile f16 [d][key]
  __shared__ __align__(16) _Float16 rel_lds[128][72];      // rel-row ring (&127)
  __shared__ __align__(16) _Float16 P_lds[WAVES][16][64];  // q·rel window (&63)
  __shared__ __align__(16) _Float16 p_lds[WAVES][16][40];  // probs staging

  const int tid  = threadIdx.x;
  const int wave = tid >> 5;
  const int lane = tid & 31;
  const int col  = lane & 15;
  const int hi   = lane >> 4;

  const int b     = blockIdx.x >> 5;
  const int qbase = (blockIdx.x & 31) * QROWS_PER_BLOCK;
  const int i0    = qbase + wave * 16;

  const float* qb = q + ((size_t)b * S_LEN) * D_ATT;
  const float4* kp4   = (const float4*)(k + ((size_t)b * S_LEN) * D_ATT);
  const float4* vp4   = (const float4*)(v + ((size_t)b * S_LEN) * D_ATT);
  const float4* relp4 = (const float4*)rel;

  const int srow = tid >> 4;   // staging rows: srow, +8, +16, +24
  const int sc4  = tid & 15;   // 16B chunk within a 64-float row

  // ---- Q tile -> two f16 A operands, pre-scaled by (1/sqrt(D))*log2(e).
  v16h qA0, qA1;
  {
    const float* qrow = qb + (size_t)(i0 + col) * D_ATT;
    const float sc = 0.125f * 1.4426950408889634f;
#pragma unroll
    for (int e = 0; e < 8; ++e) {
      qA0[e]     = (_Float16)(qrow[hi * 16 + e] * sc);
      qA0[e + 8] = (_Float16)(qrow[hi * 16 + 8 + e] * sc);
      qA1[e]     = (_Float16)(qrow[32 + hi * 16 + e] * sc);
      qA1[e + 8] = (_Float16)(qrow[32 + hi * 16 + 8 + e] * sc);
    }
  }

  const int C0  = 1024 - qbase;
  const int bw0 = 1024 - i0;

  // ---- Async DMA bases (loop-invariant LDS base; rolling global base).
  const unsigned k32base =
      (unsigned)(unsigned long long)&k32_lds[srow][sc4 * 4];
  unsigned long long kg =
      (unsigned long long)(const void*)(kp4 + (size_t)srow * 16 + sc4);

  // ---- Prologue: async-DMA K tile 0 (own chunks; waited at iter 0).
  async_cp16_off<0>(k32base, kg);
  async_cp16_off<2048>(k32base, kg);
  async_cp16_off<4096>(k32base, kg);
  async_cp16_off<6144>(k32base, kg);

  // ---- Pipeline registers: V tile 0 + rel rows [C0, C0+32).
  float4 vR[4], rR[4];
#pragma unroll
  for (int rr = 0; rr < 4; ++rr) {
    int row = srow + rr * 8;
    vR[rr] = vp4[(size_t)row * 16 + sc4];
    int t = C0 + row;
    int sr = t < 0 ? 0 : (t > 2048 ? 2048 : t);
    rR[rr] = relp4[(size_t)sr * 16 + sc4];
  }

  // ---- Prologue: stage rel rows [C0-64, C0) into the ring.
#pragma unroll 2
  for (int i = tid; i < 64 * 16; i += 128) {
    int rr = i >> 4, c4 = i & 15;
    int t = (C0 - 64) + rr;
    int sr = t < 0 ? 0 : (t > 2048 ? 2048 : t);
    *(v4h*)&rel_lds[t & 127][c4 * 4] = cvt4(relp4[(size_t)sr * 16 + c4]);
  }
  __syncthreads();

  // ---- Prologue bias block: P columns [bw0-16, bw0).
  {
    int a = bw0 - 16;
    const _Float16* rrow = rel_lds[(a + col) & 127];
    v8f pb = {};
    pb = wmma_f16(qA0, ld16(rrow + hi * 16), pb);
    pb = wmma_f16(qA1, ld16(rrow + 32 + hi * 16), pb);
#pragma unroll
    for (int r = 0; r < 8; ++r)
      P_lds[wave][r + 8 * hi][(a + col) & 63] = (_Float16)pb[r];
  }

  // ---- Flash state (C-layout striping: M = r + 8*hi).
  v8f O0 = {}, O1 = {}, O2 = {}, O3 = {};
  float rmax[8], lsum[8];
#pragma unroll
  for (int r = 0; r < 8; ++r) { rmax[r] = -3.0e38f; lsum[r] = 0.0f; }

  for (int it = 0; it < NIT; ++it) {
    const int j0  = it * 32;
    const int par = it & 1;

    // ---- Stage V^T / rel for tile `it` from pipeline registers.
#pragma unroll
    for (int rr = 0; rr < 4; ++rr) {
      int row = srow + rr * 8;
      int d0 = sc4 * 4;
      vT_lds[par][d0 + 0][row] = (_Float16)vR[rr].x;
      vT_lds[par][d0 + 1][row] = (_Float16)vR[rr].y;
      vT_lds[par][d0 + 2][row] = (_Float16)vR[rr].z;
      vT_lds[par][d0 + 3][row] = (_Float16)vR[rr].w;
      int t = C0 + j0 + row;
      *(v4h*)&rel_lds[t & 127][sc4 * 4] = cvt4(rR[rr]);
    }
    // ---- Issue V/rel global loads for tile `it+1` (hidden behind compute).
    if (it + 1 < NIT) {
      const int j1 = j0 + 32;
#pragma unroll
      for (int rr = 0; rr < 4; ++rr) {
        int row = srow + rr * 8;
        vR[rr] = vp4[(size_t)(j1 + row) * 16 + sc4];
        int t = C0 + j1 + row;
        int sr = t < 0 ? 0 : (t > 2048 ? 2048 : t);
        rR[rr] = relp4[(size_t)sr * 16 + sc4];
      }
    }

    wait_async0();  // own K DMA chunks for tile `it` have landed in k32

    // ---- Convert own DMA'd chunks f32 -> f16 into the double buffer.
#pragma unroll
    for (int rr = 0; rr < 4; ++rr) {
      int row = srow + rr * 8;
      float4 f = *(const float4*)&k32_lds[row][sc4 * 4];
      *(v4h*)&k16_lds[par][row][sc4 * 4] = cvt4(f);
    }

    __syncthreads();  // publish k16/vT/rel for tile `it` to all waves

    // ---- Async-DMA K tile `it+1` into k32. Own chunks only; the convert's
    // ds_loads already completed (register dependency), so no WAR on k32.
    if (it + 1 < NIT) {
      kg += 32 * 256;  // next 32 key rows
      async_cp16_off<0>(k32base, kg);
      async_cp16_off<2048>(k32base, kg);
      async_cp16_off<4096>(k32base, kg);
      async_cp16_off<6144>(k32base, kg);
    }

    const int bw = j0 + bw0;

    // ---- Two new bias blocks: P cols [bw, bw+16) and [bw+16, bw+32).
#pragma unroll
    for (int blk = 0; blk < 2; ++blk) {
      int a = bw + blk * 16;
      const _Float16* rrow = rel_lds[(a + col) & 127];
      v8f pb = {};
      pb = wmma_f16(qA0, ld16(rrow + hi * 16), pb);
      pb = wmma_f16(qA1, ld16(rrow + 32 + hi * 16), pb);
#pragma unroll
      for (int r = 0; r < 8; ++r)
        P_lds[wave][r + 8 * hi][(a + col) & 63] = (_Float16)pb[r];
    }

    // ---- Scores: two 16x16 tiles over 32 keys (K=64 via two WMMAs each).
    v8f sA = {}, sB = {};
    {
      const _Float16* kr0 = &k16_lds[par][col][hi * 16];
      sA = wmma_f16(qA0, ld16(kr0), sA);
      sA = wmma_f16(qA1, ld16(kr0 + 32), sA);
      const _Float16* kr1 = &k16_lds[par][16 + col][hi * 16];
      sB = wmma_f16(qA0, ld16(kr1), sB);
      sB = wmma_f16(qA1, ld16(kr1 + 32), sB);
    }

    // ---- Add relative bias via diagonal gather from the P ring.
    float sc0[8], sc1[8];
#pragma unroll
    for (int r = 0; r < 8; ++r) {
      int M = r + 8 * hi;
      sc0[r] = sA[r] + (float)P_lds[wave][M][(bw + col - M) & 63];
      sc1[r] = sB[r] + (float)P_lds[wave][M][(bw + 16 + col - M) & 63];
    }

    // ---- Online softmax in exp2 domain.
    float mx[8];
#pragma unroll
    for (int r = 0; r < 8; ++r) mx[r] = fmaxf(sc0[r], sc1[r]);
    rowmax16(mx);

    float al[8], psum[8];
#pragma unroll
    for (int r = 0; r < 8; ++r) {
      float nm = fmaxf(rmax[r], mx[r]);
      al[r] = EXP2F(rmax[r] - nm);
      rmax[r] = nm;
      float p0 = EXP2F(sc0[r] - nm);
      float p1 = EXP2F(sc1[r] - nm);
      int M = r + 8 * hi;
      p_lds[wave][M][col]      = (_Float16)p0;
      p_lds[wave][M][16 + col] = (_Float16)p1;
      psum[r] = p0 + p1;
    }
    rowsum16(psum);
#pragma unroll
    for (int r = 0; r < 8; ++r) {
      lsum[r] = lsum[r] * al[r] + psum[r];
      O0[r] *= al[r]; O1[r] *= al[r]; O2[r] *= al[r]; O3[r] *= al[r];
    }

    // ---- Probs back in A layout (16x32 over this iteration's 32 keys).
    v16h pA;
    {
      const _Float16* prow = p_lds[wave][col];
      v8h plo = *(const v8h*)(prow + hi * 16);
      v8h phi = *(const v8h*)(prow + hi * 16 + 8);
#pragma unroll
      for (int e = 0; e < 8; ++e) { pA[e] = plo[e]; pA[e + 8] = phi[e]; }
    }
    // ---- O += P · V.
    O0 = wmma_f16(pA, ld16(&vT_lds[par][col][hi * 16]), O0);
    O1 = wmma_f16(pA, ld16(&vT_lds[par][16 + col][hi * 16]), O1);
    O2 = wmma_f16(pA, ld16(&vT_lds[par][32 + col][hi * 16]), O2);
    O3 = wmma_f16(pA, ld16(&vT_lds[par][48 + col][hi * 16]), O3);
    // No trailing barrier: next iteration's staging targets the other f16
    // buffers and disjoint (mod 128) rel-ring slots; the K DMA for the next
    // tile only touches per-thread-private k32 chunks.
  }

  // ---- Epilogue: normalize and store.
  float* orow = out + ((size_t)b * S_LEN + i0) * D_ATT;
#pragma unroll
  for (int r = 0; r < 8; ++r) {
    int M = r + 8 * hi;
    float inv = 1.0f / lsum[r];
    orow[(size_t)M * D_ATT +      col] = O0[r] * inv;
    orow[(size_t)M * D_ATT + 16 + col] = O1[r] * inv;
    orow[(size_t)M * D_ATT + 32 + col] = O2[r] * inv;
    orow[(size_t)M * D_ATT + 48 + col] = O3[r] * inv;
  }
}

extern "C" void kernel_launch(void* const* d_in, const int* in_sizes, int n_in,
                              void* d_out, int out_size, void* d_ws, size_t ws_size,
                              hipStream_t stream) {
  const float* q   = (const float*)d_in[0];
  const float* k   = (const float*)d_in[1];
  const float* v   = (const float*)d_in[2];
  const float* rel = (const float*)d_in[3];
  float* out = (float*)d_out;
  (void)in_sizes; (void)n_in; (void)out_size; (void)d_ws; (void)ws_size;
  dim3 grid(BATCH * (S_LEN / QROWS_PER_BLOCK));  // 256 blocks
  dim3 block(128);                               // 4 waves (wave32)
  rga_flash_kernel<<<grid, block, 0, stream>>>(q, k, v, rel, out);
}
